// MPNN_26079041421592
// MI455X (gfx1250) — compile-verified, matrix operand order checked by reading, not run on hardware
//
#include <hip/hip_runtime.h>
#include <hip/hip_bf16.h>

// ---------------------------------------------------------------------------
// MPNN / GINEConv forward for gfx1250 (MI455X).
// Dense math via v_wmma_f32_16x16x32_bf16 (fp32 accumulate), irregular parts
// (segment-sum scatter, batchnorm) via fp32 VALU + global float atomics.
//
// v2: B weight tile stored in LDS pre-swizzled in WMMA-fragment order so each
//     lane reads its 16 bf16 B values as one aligned 32B LDS access; A rows
//     loaded as float4 chunks (the 16-bit A layout is two runs of 8
//     consecutive k per lane); K/N/strides are template constants so all
//     bounds and concat-segment checks constant-fold.
// ---------------------------------------------------------------------------

typedef __attribute__((ext_vector_type(16))) __bf16 v16bf;
typedef __attribute__((ext_vector_type(8)))  float  v8f;
typedef __attribute__((ext_vector_type(4)))  float  v4f;

#define NNODE 50000
#define NEDGE 500000
#define FIN   64
#define EIN   16
#define HDIM  100
#define WPB   4      // waves per block in the GEMM kernel

// Base K (even) held by uint v (0..7) of a 16-bit WMMA fragment (lanes 0-15;
// add 8 for lanes 16-31), per CDNA5 ISA "16-bit A-Matrix 16x32" layout.
__device__ __forceinline__ int wmma_kbase(int v) {
    return (v < 4) ? (v << 1) : (16 + ((v - 4) << 1));
}

struct GArgs {
    const float* A0;    // direct / sum operand 0 / gather table (h)
    const float* A1;    // sum operand 1 (agg)
    const float* A2;    // concat third segment (e)
    const int*   gsrc;  // gather indices (src)
    const int*   gdst;  // gather indices (dst)
    const float* W;     // weights, row-major [K][NOUT], fp32
    const float* bias;  // [NOUT]
    float*       Out;   // [M][LDC]
    float*       Acc;   // epilogue-add target (e) for EPI==2
    int          M;     // rows (multiple of 16)
};

// AMODE: 0 = A0[m*LDA+k]
//        1 = A0[m*LDA+k] + A1[m*LDA+k]                 (h + agg)
//        2 = cat( h[src[m]], h[dst[m]], A2[m] )        (edge MLP input)
//        3 = cat( relu(h[src[m]]), relu(h[dst[m]]), A2[m] )   (readout input)
// EPI:   0 = Out = acc + b ; 1 = relu ; 2 = Out = Acc + 0.5*(acc + b)
template <int AMODE, int EPI, int NT, int KS, int K_, int NOUT, int LDA, int LDC>
__global__ __launch_bounds__(WPB * 32) void gemm_wmma_kernel(GArgs g) {
    // B tile, fragment-swizzled: [ntile][lane][uint v] ; uint = 2 packed bf16.
    __shared__ __align__(32) unsigned int ldsB[NT * 32 * 8];

    const int tid    = threadIdx.x;
    const int lane   = tid & 31;
    const int wave   = tid >> 5;
    const int half   = lane >> 4;
    const int nlane  = lane & 15;
    const int mtile  = blockIdx.x * WPB + wave;
    const int m0     = mtile * 16;
    const bool active = (m0 < g.M);          // wave-uniform: M % 16 == 0
    const int mrow   = m0 + nlane;           // A row for this lane

    const float* pS = nullptr;
    const float* pD = nullptr;
    const float* pE = nullptr;
    if (AMODE <= 1) {
        int r = active ? mrow : 0;
        pS = g.A0 + (size_t)r * LDA;
        if (AMODE == 1) pD = g.A1 + (size_t)r * LDA;
    } else {
        int rs = active ? g.gsrc[mrow] : 0;
        int rd = active ? g.gdst[mrow] : 0;
        int rm = active ? mrow : 0;
        pS = g.A0 + (size_t)rs * HDIM;
        pD = g.A0 + (size_t)rd * HDIM;
        pE = g.A2 + (size_t)rm * HDIM;
    }

    v8f acc[NT];
#pragma unroll
    for (int t = 0; t < NT; ++t) acc[t] = (v8f){0.f, 0.f, 0.f, 0.f, 0.f, 0.f, 0.f, 0.f};

#pragma unroll
    for (int ks = 0; ks < KS; ++ks) {
        const int k0 = ks * 32;

        __syncthreads();  // protect LDS reuse from previous k-step
        // Cooperative fragment-swizzled fill (2 bf16 per u32 store).
#pragma unroll
        for (int it = 0; it < (NT * 32 * 8 + WPB * 32 - 1) / (WPB * 32); ++it) {
            int idx = tid + it * (WPB * 32);
            if (idx < NT * 32 * 8) {
                int t      = idx >> 8;           // /256
                int lane_s = (idx >> 3) & 31;
                int v      = idx & 7;
                int kk     = k0 + wmma_kbase(v) + ((lane_s >> 4) << 3);
                int nn     = (t << 4) + (lane_s & 15);
                float w0 = (kk < K_ && nn < NOUT) ? g.W[(size_t)kk * NOUT + nn] : 0.f;
                float w1 = ((kk + 1) < K_ && nn < NOUT) ? g.W[(size_t)(kk + 1) * NOUT + nn] : 0.f;
                union { __bf16 h[2]; unsigned int u; } pk;
                pk.h[0] = (__bf16)w0;
                pk.h[1] = (__bf16)w1;
                ldsB[idx] = pk.u;
            }
        }
        __syncthreads();

        // ---- A fragment: two runs of 8 consecutive k (chunk c: k in
        //      [k0+16c+8*half, +8) ), vectorized when the 16-wide chunk is
        //      fully in-range and within one concat segment (compile-time).
        float av[16];
#pragma unroll
        for (int c = 0; c < 2; ++c) {
            const int kc = k0 + c * 16;          // compile-time constant
            float* o = av + c * 8;
            if (kc >= ((AMODE >= 2) ? 3 * HDIM : K_)) {
#pragma unroll
                for (int j = 0; j < 8; ++j) o[j] = 0.f;
            } else if (!active) {
#pragma unroll
                for (int j = 0; j < 8; ++j) o[j] = 0.f;
            } else if (AMODE <= 1) {
                if (kc + 16 <= K_) {
                    const float* p = pS + kc + half * 8;
                    v4f q0 = *(const v4f*)(p);
                    v4f q1 = *(const v4f*)(p + 4);
                    if (AMODE == 1) {
                        const float* pd = pD + kc + half * 8;
                        v4f r0 = *(const v4f*)(pd);
                        v4f r1 = *(const v4f*)(pd + 4);
                        q0 += r0; q1 += r1;
                    }
#pragma unroll
                    for (int j = 0; j < 4; ++j) { o[j] = q0[j]; o[4 + j] = q1[j]; }
                } else {
#pragma unroll
                    for (int j = 0; j < 8; ++j) {
                        int k = kc + half * 8 + j;
                        float x = (k < K_) ? pS[k] : 0.f;
                        if (AMODE == 1 && k < K_) x += pD[k];
                        o[j] = x;
                    }
                }
            } else {  // gather-concat modes
                const int seg = kc / HDIM;
                if ((kc + 16 <= 3 * HDIM) && ((kc % HDIM) + 16 <= HDIM)) {
                    const float* sp = (seg == 0) ? pS : (seg == 1) ? pD : pE;
                    const float* p  = sp + (kc - seg * HDIM) + half * 8;
                    v4f q0 = *(const v4f*)(p);
                    v4f q1 = *(const v4f*)(p + 4);
                    if (AMODE == 3 && seg < 2) {
#pragma unroll
                        for (int j = 0; j < 4; ++j) {
                            q0[j] = fmaxf(q0[j], 0.f);
                            q1[j] = fmaxf(q1[j], 0.f);
                        }
                    }
#pragma unroll
                    for (int j = 0; j < 4; ++j) { o[j] = q0[j]; o[4 + j] = q1[j]; }
                } else {
#pragma unroll
                    for (int j = 0; j < 8; ++j) {
                        int k = kc + half * 8 + j;
                        float x = 0.f;
                        if (k < HDIM) {
                            x = pS[k];
                            if (AMODE == 3) x = fmaxf(x, 0.f);
                        } else if (k < 2 * HDIM) {
                            x = pD[k - HDIM];
                            if (AMODE == 3) x = fmaxf(x, 0.f);
                        } else if (k < 3 * HDIM) {
                            x = pE[k - 2 * HDIM];
                        }
                        o[j] = x;
                    }
                }
            }
        }
        v16bf a;
#pragma unroll
        for (int i = 0; i < 16; ++i) a[i] = (__bf16)av[i];

#pragma unroll
        for (int t = 0; t < NT; ++t) {
            const v16bf b = *reinterpret_cast<const v16bf*>(&ldsB[(t * 32 + lane) * 8]);
            acc[t] = __builtin_amdgcn_wmma_f32_16x16x32_bf16(
                false, a, false, b, (short)0, acc[t], false, false);
        }
    }

    // Epilogue: acc[t][r] holds D[m0 + r + 8*half][16t + nlane].
    if (active) {
#pragma unroll
        for (int t = 0; t < NT; ++t) {
            int n = t * 16 + nlane;
            if (t * 16 + 15 >= NOUT && n >= NOUT) continue;
            float bv = g.bias[n];
#pragma unroll
            for (int r = 0; r < 8; ++r) {
                int m = m0 + r + 8 * half;
                float  v = acc[t][r] + bv;
                size_t o = (size_t)m * LDC + n;
                if (EPI == 1) v = fmaxf(v, 0.f);
                if (EPI == 2)
                    g.Out[o] = g.Acc[o] + 0.5f * v;
                else
                    g.Out[o] = v;
            }
        }
    }
}

// ---------------------------------------------------------------------------
__global__ void zero_kernel(float* p, int n) {
    int i = blockIdx.x * blockDim.x + threadIdx.x;
    int s = gridDim.x * blockDim.x;
    for (; i < n; i += s) p[i] = 0.f;
}

// msg = relu(h[src] + e), scatter-add into agg[dst]
__global__ void msg_agg_kernel(const float* __restrict__ h, const float* __restrict__ e,
                               const int* __restrict__ src, const int* __restrict__ dst,
                               float* __restrict__ agg) {
    const int total = NEDGE * HDIM;
    int i = blockIdx.x * blockDim.x + threadIdx.x;
    int s = gridDim.x * blockDim.x;
    for (; i < total; i += s) {
        int   ed = i / HDIM;
        int   j  = i - ed * HDIM;
        float v  = h[src[ed] * HDIM + j] + e[i];
        v        = fmaxf(v, 0.f);
        atomicAdd(&agg[dst[ed] * HDIM + j], v);
    }
}

// One block per hidden column: fused BN scale/shift from batch stats.
__global__ void bn_stats_kernel(const float* __restrict__ z, float* __restrict__ scale,
                                float* __restrict__ shift, const float* __restrict__ gamma,
                                const float* __restrict__ beta) {
    int j = blockIdx.x;
    float s1 = 0.f, s2 = 0.f;
    for (int i = threadIdx.x; i < NNODE; i += blockDim.x) {
        float v = z[i * HDIM + j];
        s1 += v;
        s2 += v * v;
    }
    __shared__ float a1[256], a2[256];
    a1[threadIdx.x] = s1;
    a2[threadIdx.x] = s2;
    __syncthreads();
    for (int st = 128; st > 0; st >>= 1) {
        if (threadIdx.x < st) {
            a1[threadIdx.x] += a1[threadIdx.x + st];
            a2[threadIdx.x] += a2[threadIdx.x + st];
        }
        __syncthreads();
    }
    if (threadIdx.x == 0) {
        float mu  = a1[0] / (float)NNODE;
        float var = a2[0] / (float)NNODE - mu * mu;
        float sc  = rsqrtf(var + 1e-5f) * gamma[j];
        scale[j]  = sc;
        shift[j]  = beta[j] - mu * sc;
    }
}

// h = (h + relu(z*scale + shift)) * 0.5
__global__ void bn_apply_kernel(float* __restrict__ h, const float* __restrict__ z,
                                const float* __restrict__ scale, const float* __restrict__ shift) {
    const int total = NNODE * HDIM;
    int i = blockIdx.x * blockDim.x + threadIdx.x;
    int s = gridDim.x * blockDim.x;
    for (; i < total; i += s) {
        int   j  = i % HDIM;
        float zn = z[i] * scale[j] + shift[j];
        h[i]     = (h[i] + fmaxf(zn, 0.f)) * 0.5f;
    }
}

// out[E,2] = o2[E,25] @ w3[25,2] + b3     (o2 has padded row stride 32)
__global__ void final_out_kernel(const float* __restrict__ o2, const float* __restrict__ w3,
                                 const float* __restrict__ b3, float* __restrict__ out) {
    int ed = blockIdx.x * blockDim.x + threadIdx.x;
    if (ed >= NEDGE) return;
    float a0 = b3[0], a1 = b3[1];
#pragma unroll
    for (int k = 0; k < 25; ++k) {
        float v = o2[ed * 32 + k];
        a0 += v * w3[k * 2 + 0];
        a1 += v * w3[k * 2 + 1];
    }
    out[ed * 2 + 0] = a0;
    out[ed * 2 + 1] = a1;
}

// ---------------------------------------------------------------------------
static inline int gemm_blocks(int M) {
    int mtiles = (M + 15) / 16;
    return (mtiles + WPB - 1) / WPB;
}

extern "C" void kernel_launch(void* const* d_in, const int* in_sizes, int n_in,
                              void* d_out, int out_size, void* d_ws, size_t ws_size,
                              hipStream_t stream) {
    const float* x         = (const float*)d_in[0];
    const float* edge_attr = (const float*)d_in[1];
    const int*   ei        = (const int*)d_in[2];
    const float* node_w    = (const float*)d_in[3];
    const float* node_b    = (const float*)d_in[4];
    const float* edge_w    = (const float*)d_in[5];
    const float* edge_b    = (const float*)d_in[6];
    const float* gine_w1   = (const float*)d_in[7];
    const float* gine_b1   = (const float*)d_in[8];
    const float* gine_w2   = (const float*)d_in[9];
    const float* gine_b2   = (const float*)d_in[10];
    const float* bn_gamma  = (const float*)d_in[11];
    const float* bn_beta   = (const float*)d_in[12];
    const float* emlp_w1   = (const float*)d_in[13];
    const float* emlp_b1   = (const float*)d_in[14];
    const float* emlp_w2   = (const float*)d_in[15];
    const float* emlp_b2   = (const float*)d_in[16];
    const float* mlp_w1    = (const float*)d_in[17];
    const float* mlp_b1    = (const float*)d_in[18];
    const float* mlp_w2    = (const float*)d_in[19];
    const float* mlp_b2    = (const float*)d_in[20];
    const float* mlp_w3    = (const float*)d_in[21];
    const float* mlp_b3    = (const float*)d_in[22];

    const int* src = ei;
    const int* dst = ei + NEDGE;

    // Workspace carve-up (floats).
    float*     ws = (float*)d_ws;
    const int  NH = NNODE * HDIM;  // 5,000,000
    const int  EH = NEDGE * HDIM;  // 50,000,000
    float* h     = ws;
    float* agg   = h + NH;
    float* z     = agg + NH;
    float* tn    = z + NH;
    float* stats = tn + NH;                    // scale[0:100], shift at +128
    float* e     = stats + 256;
    float* te    = e + EH;                     // edge hidden; readout o1/o2 too
    float* o1    = te;                         // [E, 50] stride 64
    float* o2    = te + (size_t)NEDGE * 64;    // [E, 25] stride 32
    float* scale = stats;
    float* shift = stats + 128;

    dim3 blk(WPB * 32);
    const int ew_blocks = 2048;
    (void)n_in; (void)in_sizes; (void)out_size; (void)ws_size;

    // 1) h = x @ node_w + node_b                     [50000,64]x[64,100]
    {
        GArgs g{x, nullptr, nullptr, nullptr, nullptr, node_w, node_b, h, nullptr, NNODE};
        gemm_wmma_kernel<0, 0, 7, 2, 64, 100, 64, 100>
            <<<gemm_blocks(NNODE), blk, 0, stream>>>(g);
    }
    // 2) e = edge_attr @ edge_w + edge_b             [500000,16]x[16,100]
    {
        GArgs g{edge_attr, nullptr, nullptr, nullptr, nullptr, edge_w, edge_b, e, nullptr, NEDGE};
        gemm_wmma_kernel<0, 0, 7, 1, 16, 100, 16, 100>
            <<<gemm_blocks(NEDGE), blk, 0, stream>>>(g);
    }

    for (int layer = 0; layer < 2; ++layer) {
        const float* gw1 = gine_w1 + layer * HDIM * HDIM;
        const float* gb1 = gine_b1 + layer * HDIM;
        const float* gw2 = gine_w2 + layer * HDIM * HDIM;
        const float* gb2 = gine_b2 + layer * HDIM;
        const float* gga = bn_gamma + layer * HDIM;
        const float* gbe = bn_beta + layer * HDIM;
        const float* ew1 = emlp_w1 + layer * 3 * HDIM * HDIM;
        const float* eb1 = emlp_b1 + layer * HDIM;
        const float* ew2 = emlp_w2 + layer * HDIM * HDIM;
        const float* eb2 = emlp_b2 + layer * HDIM;

        // a) agg = segment_sum(relu(h[src]+e), dst)
        zero_kernel<<<ew_blocks, 256, 0, stream>>>(agg, NH);
        msg_agg_kernel<<<ew_blocks, 256, 0, stream>>>(h, e, src, dst, agg);

        // b) tn = relu((h+agg) @ gw1 + gb1) ; z = tn @ gw2 + gb2
        {
            GArgs g{h, agg, nullptr, nullptr, nullptr, gw1, gb1, tn, nullptr, NNODE};
            gemm_wmma_kernel<1, 1, 7, 4, 100, 100, 100, 100>
                <<<gemm_blocks(NNODE), blk, 0, stream>>>(g);
        }
        {
            GArgs g{tn, nullptr, nullptr, nullptr, nullptr, gw2, gb2, z, nullptr, NNODE};
            gemm_wmma_kernel<0, 0, 7, 4, 100, 100, 100, 100>
                <<<gemm_blocks(NNODE), blk, 0, stream>>>(g);
        }

        // c) batchnorm (batch stats) + residual update of h
        bn_stats_kernel<<<HDIM, 256, 0, stream>>>(z, scale, shift, gga, gbe);
        bn_apply_kernel<<<ew_blocks, 256, 0, stream>>>(h, z, scale, shift);

        // d) edge update: e += 0.5 * MLP(cat(h[src], h[dst], e))
        {
            GArgs g{h, nullptr, e, src, dst, ew1, eb1, te, nullptr, NEDGE};
            gemm_wmma_kernel<2, 1, 7, 10, 300, 100, 100, 100>
                <<<gemm_blocks(NEDGE), blk, 0, stream>>>(g);
        }
        {
            GArgs g{te, nullptr, nullptr, nullptr, nullptr, ew2, eb2, e, e, NEDGE};
            gemm_wmma_kernel<0, 2, 7, 4, 100, 100, 100, 100>
                <<<gemm_blocks(NEDGE), blk, 0, stream>>>(g);
        }
    }

    // Readout: o1 = relu(cat(relu(h[src]),relu(h[dst]),e) @ mlp_w1 + b1)  [E,300]x[300,50]
    {
        GArgs g{h, nullptr, e, src, dst, mlp_w1, mlp_b1, o1, nullptr, NEDGE};
        gemm_wmma_kernel<3, 1, 4, 10, 300, 50, 100, 64>
            <<<gemm_blocks(NEDGE), blk, 0, stream>>>(g);
    }
    // o2 = relu(o1 @ mlp_w2 + b2)                                         [E,50]x[50,25]
    {
        GArgs g{o1, nullptr, nullptr, nullptr, nullptr, mlp_w2, mlp_b2, o2, nullptr, NEDGE};
        gemm_wmma_kernel<0, 1, 2, 2, 50, 25, 64, 32>
            <<<gemm_blocks(NEDGE), blk, 0, stream>>>(g);
    }
    // out = o2 @ mlp_w3 + b3                                              [E,25]x[25,2]
    final_out_kernel<<<(NEDGE + 255) / 256, 256, 0, stream>>>(o2, mlp_w3, mlp_b3,
                                                              (float*)d_out);
}
